// BinHD_32873679684016
// MI455X (gfx1250) — 8-bit matrix operand order repaired and checked by reading, compiled, not run on hardware
//
#include <hip/hip_runtime.h>

typedef __attribute__((ext_vector_type(16))) int   v16i;
typedef __attribute__((ext_vector_type(8)))  float v8f;

#define N_SAMPLES 8192
#define N_CLASSES 1000
#define C_PAD     1024
#define DIM       10000
#define D_PAD     10112

#define FP8_ONE   0x38

__global__ __launch_bounds__(256) void bhd_convert_fp8(
    const float* __restrict__ src, unsigned char* __restrict__ dst,
    float* __restrict__ sums, int valid_rows)
{
    const int row = blockIdx.x;
    const int tid = threadIdx.x;
    __shared__ float red[256];

    float acc = 0.0f;
    uchar4* d4 = (uchar4*)(dst + (size_t)row * D_PAD);

    if (row < valid_rows) {
        const float4* s4 = (const float4*)(src + (size_t)row * DIM);
        for (int i = tid; i < DIM / 4; i += 256) {
            float4 v = s4[i];
            uchar4 o;
            o.x = (v.x > 0.5f) ? (unsigned char)FP8_ONE : (unsigned char)0;
            o.y = (v.y > 0.5f) ? (unsigned char)FP8_ONE : (unsigned char)0;
            o.z = (v.z > 0.5f) ? (unsigned char)FP8_ONE : (unsigned char)0;
            o.w = (v.w > 0.5f) ? (unsigned char)FP8_ONE : (unsigned char)0;
            d4[i] = o;
            acc += v.x + v.y + v.z + v.w;
        }
        for (int i = DIM / 4 + tid; i < D_PAD / 4; i += 256)
            d4[i] = make_uchar4(0, 0, 0, 0);
    } else {
        for (int i = tid; i < D_PAD / 4; i += 256)
            d4[i] = make_uchar4(0, 0, 0, 0);
    }

    red[tid] = acc;
    __syncthreads();
    for (int s = 128; s > 0; s >>= 1) {
        if (tid < s) red[tid] += red[tid + s];
        __syncthreads();
    }
    if (tid == 0) sums[row] = red[0];
}

__global__ __launch_bounds__(256) void bhd_gemm_fp8(
    const unsigned char* __restrict__ A,
    const unsigned char* __restrict__ Bp,
    const float* __restrict__ sSum,
    const float* __restrict__ cSum,
    float* __restrict__ out)
{
    const int lane  = threadIdx.x & 31;
    const int wave  = threadIdx.x >> 5;
    const int half  = lane >> 4;
    const int lr    = lane & 15;
    const int mBase = blockIdx.y * 256 + wave * 32;
    const int nBase = blockIdx.x * 64;

    const unsigned char* aRow0 = A + (size_t)(mBase +  0 + lr) * D_PAD + half * 16;
    const unsigned char* aRow1 = A + (size_t)(mBase + 16 + lr) * D_PAD + half * 16;
    const unsigned char* bRow0 = Bp + (size_t)(nBase +  0 + lr) * D_PAD + half * 16;
    const unsigned char* bRow1 = Bp + (size_t)(nBase + 16 + lr) * D_PAD + half * 16;
    const unsigned char* bRow2 = Bp + (size_t)(nBase + 32 + lr) * D_PAD + half * 16;
    const unsigned char* bRow3 = Bp + (size_t)(nBase + 48 + lr) * D_PAD + half * 16;

    v8f acc00 = {}, acc01 = {}, acc02 = {}, acc03 = {};
    v8f acc10 = {}, acc11 = {}, acc12 = {}, acc13 = {};

    for (int k = 0; k < D_PAD; k += 128) {
        v16i a0, a1;
        {
            const unsigned char* p0 = aRow0 + k;
            const unsigned char* p1 = aRow1 + k;
#pragma unroll
            for (int c = 0; c < 8; ++c) {
                uint2 q0 = *(const uint2*)(p0 + 32 * (c >> 1) + 8 * (c & 1));
                uint2 q1 = *(const uint2*)(p1 + 32 * (c >> 1) + 8 * (c & 1));
                a0[2 * c + 0] = (int)q0.x;
                a0[2 * c + 1] = (int)q0.y;
                a1[2 * c + 0] = (int)q1.x;
                a1[2 * c + 1] = (int)q1.y;
            }
        }

#define BHD_TILE(ACC0, ACC1, BROW)                                              \
        {                                                                       \
            const unsigned char* p = (BROW) + k;                                \
            uint4 q0 = *(const uint4*)(p +  0);                                 \
            uint4 q1 = *(const uint4*)(p + 32);                                 \
            uint4 q2 = *(const uint4*)(p + 64);                                 \
            uint4 q3 = *(const uint4*)(p + 96);                                 \
            v16i b;                                                             \
            b[0]  = (int)q0.x; b[1]  = (int)q0.y; b[2]  = (int)q0.z; b[3]  = (int)q0.w; \
            b[4]  = (int)q1.x; b[5]  = (int)q1.y; b[6]  = (int)q1.z; b[7]  = (int)q1.w; \
            b[8]  = (int)q2.x; b[9]  = (int)q2.y; b[10] = (int)q2.z; b[11] = (int)q2.w; \
            b[12] = (int)q3.x; b[13] = (int)q3.y; b[14] = (int)q3.z; b[15] = (int)q3.w; \
            ACC0 = __builtin_amdgcn_wmma_f32_16x16x128_fp8_fp8(                 \
                       a0, b, (short)0, ACC0, false, false);                    \
            ACC1 = __builtin_amdgcn_wmma_f32_16x16x128_fp8_fp8(                 \
                       a1, b, (short)0, ACC1, false, false);                    \
        }

        BHD_TILE(acc00, acc10, bRow0)
        BHD_TILE(acc01, acc11, bRow1)
        BHD_TILE(acc02, acc12, bRow2)
        BHD_TILE(acc03, acc13, bRow3)
#undef BHD_TILE
    }

    const int mRow0 = mBase + half * 8;
    float sv0[8], sv1[8];
#pragma unroll
    for (int j = 0; j < 8; ++j) {
        sv0[j] = sSum[mRow0 + j];
        sv1[j] = sSum[mRow0 + 16 + j];
    }

#define BHD_STORE(ACC0, ACC1, T)                                                \
    {                                                                           \
        int col = nBase + (T) * 16 + lr;                                        \
        if (col < N_CLASSES) {                                                  \
            float cv = cSum[col];                                               \
            _Pragma("unroll")                                                   \
            for (int j = 0; j < 8; ++j) {                                       \
                out[(size_t)(mRow0 + j) * N_CLASSES + col] =                    \
                    sv0[j] + cv - 2.0f * ACC0[j];                               \
                out[(size_t)(mRow0 + 16 + j) * N_CLASSES + col] =               \
                    sv1[j] + cv - 2.0f * ACC1[j];                               \
            }                                                                   \
        }                                                                       \
    }

    BHD_STORE(acc00, acc10, 0)
    BHD_STORE(acc01, acc11, 1)
    BHD_STORE(acc02, acc12, 2)
    BHD_STORE(acc03, acc13, 3)
#undef BHD_STORE
}

extern "C" void kernel_launch(void* const* d_in, const int* in_sizes, int n_in,
                              void* d_out, int out_size, void* d_ws, size_t ws_size,
                              hipStream_t stream)
{
    (void)in_sizes; (void)n_in; (void)out_size; (void)ws_size;

    const float* samples = (const float*)d_in[0];
    const float* classes = (const float*)d_in[1];
    float*       out     = (float*)d_out;

    unsigned char* ws   = (unsigned char*)d_ws;
    unsigned char* sA   = ws;
    unsigned char* sB   = sA + (size_t)N_SAMPLES * D_PAD;
    float*         sSum = (float*)(sB + (size_t)C_PAD * D_PAD);
    float*         cSum = sSum + N_SAMPLES;

    bhd_convert_fp8<<<dim3(N_SAMPLES), dim3(256), 0, stream>>>(samples, sA, sSum, N_SAMPLES);
    bhd_convert_fp8<<<dim3(C_PAD),     dim3(256), 0, stream>>>(classes, sB, cSum, N_CLASSES);

    bhd_gemm_fp8<<<dim3(C_PAD / 64, N_SAMPLES / 256), dim3(256), 0, stream>>>(
        sA, sB, sSum, cSum, out);
}
